// RoIFeatureExtractor_4818953306580
// MI455X (gfx1250) — matrix operand order repaired and checked by reading, not compiled, not run on hardware
//
#include <hip/hip_runtime.h>
#include <hip/hip_bf16.h>

typedef _Float16 v16h __attribute__((ext_vector_type(16)));
typedef _Float16 v8h  __attribute__((ext_vector_type(8)));
typedef _Float16 v4h  __attribute__((ext_vector_type(4)));
typedef float    v8f  __attribute__((ext_vector_type(8)));
typedef int      v4i  __attribute__((ext_vector_type(4)));

#define C_DIM 1024
#define H_DIM 40
#define W_DIM 40
#define HW    1600          // H*W, also the GEMM K dim (divisible by 32)
#define BM    128           // block tile M (ROIs)
#define BN    64            // block tile N (channels)
#define BK    32            // K step per WMMA chain
#define LDP   40            // padded halfs per LDS row (80B stride)

// ---------------------------------------------------------------------------
// Async global->LDS staging (CDNA5 GLOBAL_LOAD_ASYNC_TO_LDS_B128, ASYNCcnt),
// with a synchronous vector-copy fallback if the builtin isn't declared.
// Builtin params (per hipcc diagnostic): (int4 AS1* src, int4 AS3* dst, imm, imm)
// ---------------------------------------------------------------------------
#if __has_builtin(__builtin_amdgcn_global_load_async_to_lds_b128)
#define ASYNC_LDS 1
typedef __attribute__((address_space(1))) v4i* gbl_v4i;
typedef __attribute__((address_space(3))) v4i* lds_v4i;
#else
#define ASYNC_LDS 0
#endif

__device__ __forceinline__ void copy16(const _Float16* __restrict__ g,
                                       _Float16* __restrict__ l) {
#if ASYNC_LDS
    __builtin_amdgcn_global_load_async_to_lds_b128(
        (gbl_v4i)(g), (lds_v4i)(l), 0, 0);
#else
    *(v8h*)l = *(const v8h*)g;
#endif
}

__device__ __forceinline__ void async_join() {
#if ASYNC_LDS
#if __has_builtin(__builtin_amdgcn_s_wait_asynccnt)
    __builtin_amdgcn_s_wait_asynccnt(0);
#else
    asm volatile("s_wait_asynccnt 0x0" ::: "memory");
#endif
#endif
}

// ---------------------------------------------------------------------------
// Kernel 1: F2 = (f16) fm, elementwise (channel-major layout kept as-is:
// F2[c][p]; the GEMM's B[k][n] = F2[n][k] so B rows are already K-contiguous).
// ---------------------------------------------------------------------------
__global__ void roi_convert_f16(const float* __restrict__ fm,
                                _Float16* __restrict__ F2) {
    const int i = (blockIdx.x * 256 + threadIdx.x) * 4;   // total = C_DIM*HW, /4 exact
    const float4 v = *(const float4*)(fm + i);
    v4h h;
    h[0] = (_Float16)v.x; h[1] = (_Float16)v.y;
    h[2] = (_Float16)v.z; h[3] = (_Float16)v.w;
    *(v4h*)(F2 + i) = h;
}

// ---------------------------------------------------------------------------
// Kernel 2: one block per ROI. Scatter the 49 bilinear taps (4 corners each,
// weight/49) into a 1600-float LDS accumulator (ds_add_f32), emit f16 row.
// Faithful port of the reference clip/floor/frac math; invalid corners are
// skipped (reference zeroes their sampled value).
// ---------------------------------------------------------------------------
__global__ void roi_weights(const float* __restrict__ bbox,
                            const int* __restrict__ gsp,
                            const int* __restrict__ isp,
                            _Float16* __restrict__ Wh) {
    __shared__ float acc[HW];
    const int n = blockIdx.x;
    const int t = threadIdx.x;

    for (int i = t; i < HW; i += 256) acc[i] = 0.0f;
    __syncthreads();

    const float gs    = (float)gsp[0];
    const float is    = (float)isp[0];
    const float scale = (is / gs) * gs;       // == image_size, computed faithfully

    const float cx = bbox[n * 4 + 0];
    const float cy = bbox[n * 4 + 1];
    const float w  = bbox[n * 4 + 2];
    const float h  = bbox[n * 4 + 3];

    const float x1 = fminf(fmaxf((cx - w * 0.5f) / scale, 0.0f), 1.0f);
    const float x2 = fminf(fmaxf((cx + w * 0.5f) / scale, 0.0f), 1.0f);
    const float y1 = fminf(fmaxf((cy - h * 0.5f) / scale, 0.0f), 1.0f);
    const float y2 = fminf(fmaxf((cy + h * 0.5f) / scale, 0.0f), 1.0f);
    const float bw = x2 - x1, bh = y2 - y1;
    const float txc = x1 + x2 - 1.0f, tyc = y1 + y2 - 1.0f;

    if (t < 49 * 4) {
        const int k      = t >> 2;
        const int corner = t & 3;
        const int kx = k % 7, ky = k / 7;
        const float bx = (2.0f * (float)kx + 1.0f) / 7.0f - 1.0f;
        const float by = (2.0f * (float)ky + 1.0f) / 7.0f - 1.0f;
        const float ixf = (bw * bx + txc + 1.0f) * ((float)W_DIM * 0.5f) - 0.5f;
        const float iyf = (bh * by + tyc + 1.0f) * ((float)H_DIM * 0.5f) - 0.5f;
        const float x0f = floorf(ixf), y0f = floorf(iyf);
        const float fx = ixf - x0f,   fy = iyf - y0f;
        const int xi = (int)x0f + (corner & 1);
        const int yi = (int)y0f + (corner >> 1);
        const float wx = (corner & 1)  ? fx : 1.0f - fx;
        const float wy = (corner >> 1) ? fy : 1.0f - fy;
        if (xi >= 0 && xi < W_DIM && yi >= 0 && yi < H_DIM)
            atomicAdd(&acc[yi * W_DIM + xi], wx * wy * (1.0f / 49.0f));
    }
    __syncthreads();

    for (int i = t; i < HW; i += 256)
        Wh[(size_t)n * HW + i] = (_Float16)acc[i];
}

// ---------------------------------------------------------------------------
// Fragment builders (CDNA5 wave32 16-bit layouts, two b128 LDS reads each).
// A 16x32: lanes 0-15 -> M=lane, K{0..7,16..23}; lanes 16-31 -> K{8..15,24..31}
// B 32x16: lanes 0-15 -> N=lane, K{0..7,16..23}; lanes 16-31 -> K{8..15,24..31}
// ---------------------------------------------------------------------------
__device__ __forceinline__ v16h fragA(const _Float16* rowp, int kb) {
    v8h lo = *(const v8h*)(rowp + kb);
    v8h hi = *(const v8h*)(rowp + kb + 16);
    v16h a;
#pragma unroll
    for (int j = 0; j < 8; ++j) { a[j] = lo[j]; a[j + 8] = hi[j]; }
    return a;
}
__device__ __forceinline__ v16h fragB(const _Float16* rowp, int kb) {
    v8h lo = *(const v8h*)(rowp + kb);
    v8h hi = *(const v8h*)(rowp + kb + 16);
    v16h b;
#pragma unroll
    for (int j = 0; j < 8; ++j) { b[j] = lo[j]; b[j + 8] = hi[j]; }
    return b;
}

// ---------------------------------------------------------------------------
// Kernel 3: out[N x 1024] f32 = Wh[N x 1600] f16 x F2^T; 128x64 block tile,
// 8 wave32s, each wave owns a 32x32 patch (4 accumulators, 4 WMMA / K-step).
// ---------------------------------------------------------------------------
__global__ void roi_gemm_wmma(const _Float16* __restrict__ Wh,
                              const _Float16* __restrict__ F2,
                              float* __restrict__ out, int N) {
    __shared__ _Float16 ldsA[BM][LDP];   // [m][k]
    __shared__ _Float16 ldsB[BN][LDP];   // [n][k]

    const int tid  = threadIdx.x;
    const int lane = tid & 31;
    const int wid  = tid >> 5;               // 0..7
    const int mW   = (wid & 3) * 32;         // wave M offset within block tile
    const int nW   = (wid >> 2) * 32;        // wave N offset (0 or 32)
    const int m0   = blockIdx.y * BM;
    const int n0   = blockIdx.x * BN;

    v8f c00 = {}, c01 = {}, c10 = {}, c11 = {};

    const int arow = tid >> 2;               // 0..63
    const int aseg = (tid & 3) * 8;          // 16B K sub-chunk

    for (int k0 = 0; k0 < HW; k0 += BK) {
        // A tile: 128 rows x 32 k, two 16B chunks per thread (guarded M tail)
#pragma unroll
        for (int rr = 0; rr < BM; rr += 64) {
            const int r = rr + arow;
            if (m0 + r < N)
                copy16(Wh + (size_t)(m0 + r) * HW + k0 + aseg, &ldsA[r][aseg]);
            else
                *(v8h*)&ldsA[r][aseg] = (v8h){};
        }
        // B tile: 64 rows (channels) x 32 k, one 16B chunk per thread
        copy16(F2 + (size_t)(n0 + arow) * HW + k0 + aseg, &ldsB[arow][aseg]);

        async_join();
        __syncthreads();

        const int kbA = (lane < 16) ? 0 : 8;
        const int kbB = (lane < 16) ? 0 : 8;
        const int rA  = mW + (lane & 15);
        const int rB  = nW + (lane & 15);

        v16h a0 = fragA(&ldsA[rA][0],      kbA);
        v16h a1 = fragA(&ldsA[rA + 16][0], kbA);
        v16h b0 = fragB(&ldsB[rB][0],      kbB);
        v16h b1 = fragB(&ldsB[rB + 16][0], kbB);

        c00 = __builtin_amdgcn_wmma_f32_16x16x32_f16(false, a0, false, b0, (short)0, c00, false, false);
        c01 = __builtin_amdgcn_wmma_f32_16x16x32_f16(false, a0, false, b1, (short)0, c01, false, false);
        c10 = __builtin_amdgcn_wmma_f32_16x16x32_f16(false, a1, false, b0, (short)0, c10, false, false);
        c11 = __builtin_amdgcn_wmma_f32_16x16x32_f16(false, a1, false, b1, (short)0, c11, false, false);
        __syncthreads();
    }

    // C/D layout: VGPR j holds M = j (lanes 0-15) / M = j+8 (lanes 16-31), N = lane%16
    const int col = n0 + nW + (lane & 15);
    const int rb  = m0 + mW + ((lane < 16) ? 0 : 8);
#pragma unroll
    for (int j = 0; j < 8; ++j) {
        const int r0 = rb + j;
        const int r1 = rb + 16 + j;
        if (r0 < N) {
            out[(size_t)r0 * C_DIM + col]      = c00[j];
            out[(size_t)r0 * C_DIM + col + 16] = c01[j];
        }
        if (r1 < N) {
            out[(size_t)r1 * C_DIM + col]      = c10[j];
            out[(size_t)r1 * C_DIM + col + 16] = c11[j];
        }
    }
}

// ---------------------------------------------------------------------------
extern "C" void kernel_launch(void* const* d_in, const int* in_sizes, int n_in,
                              void* d_out, int out_size, void* d_ws, size_t ws_size,
                              hipStream_t stream) {
    const float* fm   = (const float*)d_in[0];   // [1024, 40, 40]
    const float* bbox = (const float*)d_in[1];   // [N, 4]
    const int*   gsp  = (const int*)d_in[2];     // grid_size
    const int*   isp  = (const int*)d_in[3];     // image_size
    float*       out  = (float*)d_out;           // [N, 1024]
    const int N = in_sizes[1] / 4;

    // workspace: F2 (1024*1600 f16 = 3.28 MB) then Wh (N*1600 f16 = 13.1 MB)
    _Float16* F2 = (_Float16*)d_ws;
    _Float16* Wh = (_Float16*)((char*)d_ws + (size_t)HW * C_DIM * sizeof(_Float16));

    roi_convert_f16<<<dim3((C_DIM * HW) / (256 * 4)), 256, 0, stream>>>(fm, F2);
    roi_weights<<<dim3(N), 256, 0, stream>>>(bbox, gsp, isp, Wh);
    roi_gemm_wmma<<<dim3(C_DIM / BN, (N + BM - 1) / BM), 256, 0, stream>>>(Wh, F2, out, N);
}
